// GatingAttention_28733331210462
// MI455X (gfx1250) — matrix operand order repaired and checked by reading, not compiled, not run on hardware
//
#include <hip/hip_runtime.h>
#include <hip/hip_bf16.h>

// ---------------------------------------------------------------------------
// Gated attention for MI455X (gfx1250, wave32, WMMA bf16 16x16x32).
// B=2, L=1024, DIM=1024, H=16, hd=64.
// All five matmuls on v_wmma_f32_16x16x32_bf16; operands pre-arranged so every
// fragment load from global is two 16B k-runs per lane (2x b128).
// ---------------------------------------------------------------------------

static constexpr int kB    = 2;
static constexpr int kL    = 1024;
static constexpr int kDIM  = 1024;
static constexpr int kH    = 16;
static constexpr int kHD   = 64;
static constexpr int kROWS = kB * kL;        // 2048
static constexpr int kQKVN = 3 * kDIM;       // 3072

typedef __attribute__((ext_vector_type(16))) __bf16 v16bf;
typedef __attribute__((ext_vector_type(8)))  __bf16 v8bf;   // 16 bytes
typedef __attribute__((ext_vector_type(8)))  float  v8f;
typedef __attribute__((ext_vector_type(4)))  float  f4;

static __device__ __forceinline__ __bf16 to_bf(float f) { return (__bf16)f; }

static __device__ __forceinline__ v8f wmma_bf16(v16bf a, v16bf b, v8f c) {
    return __builtin_amdgcn_wmma_f32_16x16x32_bf16(
        false, a, false, b, (short)0, c, false, false);
}

// A fragment: 16x32 bf16 from row-major [M x K] source.
// lane<16: K = k0+0..7 | k0+16..23 ; lane>=16: K = k0+8..15 | k0+24..31
static __device__ __forceinline__ v16bf load_a_frag(const __bf16* __restrict__ A,
                                                    int lda, int row0, int k0, int lane) {
    int r  = row0 + (lane & 15);
    int kb = k0 + ((lane & 16) ? 8 : 0);
    const v8bf* p = (const v8bf*)(A + (size_t)r * lda + kb);
    v8bf lo = p[0];      // K = kb .. kb+7
    v8bf hi = p[2];      // K = kb+16 .. kb+23
    v16bf a;
#pragma unroll
    for (int e = 0; e < 8; ++e) { a[e] = lo[e]; a[8 + e] = hi[e]; }
    return a;
}

// B fragment: 32x16 bf16 where the source is laid out contiguous along the
// WMMA k-dim with leading stride ld per output column.
// lane<16: K = k0+0..7 | k0+16..23 ; lane>=16: K = k0+8..15 | k0+24..31
static __device__ __forceinline__ v16bf load_b_frag(const __bf16* __restrict__ Bm,
                                                    int ld, int k0, int col0, int lane) {
    int c  = col0 + (lane & 15);
    int kb = k0 + ((lane & 16) ? 8 : 0);
    const v8bf* p = (const v8bf*)(Bm + (size_t)c * ld + kb);
    v8bf lo = p[0];
    v8bf hi = p[2];
    v16bf b;
#pragma unroll
    for (int e = 0; e < 8; ++e) { b[e] = lo[e]; b[8 + e] = hi[e]; }
    return b;
}

// A fragment sourced from fp32 LDS (softmax probabilities), converted to bf16.
static __device__ __forceinline__ v16bf load_a_frag_lds(const float* __restrict__ S,
                                                        int lds, int k0, int lane) {
    int r  = lane & 15;
    int kb = k0 + ((lane & 16) ? 8 : 0);
    const f4* p = (const f4*)(S + r * lds + kb);
    f4 a0 = p[0], a1 = p[1];    // K = kb..kb+7
    f4 b0 = p[4], b1 = p[5];    // K = kb+16..kb+23
    v16bf a;
#pragma unroll
    for (int e = 0; e < 4; ++e) {
        a[e]      = to_bf(a0[e]);
        a[4 + e]  = to_bf(a1[e]);
        a[8 + e]  = to_bf(b0[e]);
        a[12 + e] = to_bf(b1[e]);
    }
    return a;
}

// ---------------------------------------------------------------------------
__global__ void cvt_f32_to_bf16(const float* __restrict__ in, __bf16* __restrict__ out, int n) {
    int i = blockIdx.x * blockDim.x + threadIdx.x;
    if (i < n) out[i] = to_bf(in[i]);
}

// Transpose+convert: in [K, N] f32 -> out [N, K] bf16 (tiled via LDS).
__global__ __launch_bounds__(256) void cvt_transpose_bf16(const float* __restrict__ in,
                                                          __bf16* __restrict__ out,
                                                          int K, int N) {
    __shared__ float tile[16][17];
    int n0 = blockIdx.x * 16, k0 = blockIdx.y * 16;
    int tx = threadIdx.x & 15, ty = threadIdx.x >> 4;
    tile[ty][tx] = in[(size_t)(k0 + ty) * N + n0 + tx];
    __syncthreads();
    out[(size_t)(n0 + ty) * K + k0 + tx] = to_bf(tile[tx][ty]);
}

// ---------------------------------------------------------------------------
// QKV projection: X[2048,1024] x Wt[3072,1024]^T -> Q,K head-major [B,H,L,hd],
// V transposed [B,H,hd,L]. h / section / batch are wave-uniform per tile.
__global__ __launch_bounds__(256) void qkv_gemm(const __bf16* __restrict__ Xb,
                                                const __bf16* __restrict__ Wt,
                                                __bf16* __restrict__ Q,
                                                __bf16* __restrict__ K,
                                                __bf16* __restrict__ Vt) {
    const int lane = threadIdx.x & 31;
    const int wid  = blockIdx.x * (blockDim.x >> 5) + (threadIdx.x >> 5);
    const int tilesN = kQKVN / 64;                    // 48
    const int tm = wid / tilesN, tn = wid % tilesN;
    if (tm >= kROWS / 16) return;
    const int m0 = tm * 16, n0 = tn * 64;

    v8f acc[4] = {};
    for (int k0 = 0; k0 < kDIM; k0 += 32) {
        v16bf a = load_a_frag(Xb, kDIM, m0, k0, lane);
#pragma unroll
        for (int t = 0; t < 4; ++t)
            acc[t] = wmma_bf16(a, load_b_frag(Wt, kDIM, k0, n0 + t * 16, lane), acc[t]);
    }

    // wave-uniform destination decode
    const int bb   = m0 >> 10;           // batch
    const int l0   = m0 & (kL - 1);      // row within batch
    const int h    = n0 / (3 * kHD);     // head
    const int sect = (n0 % (3 * kHD)) >> 6;  // 0=q,1=k,2=v
    const int hi8  = (lane >> 4) << 3;
    const int cl   = lane & 15;
    const size_t bh = (size_t)bb * kH + h;

    if (sect == 2) {                     // V: transposed [bh, c, l]
        __bf16* base = Vt + bh * kHD * kL + l0;
#pragma unroll
        for (int t = 0; t < 4; ++t)
#pragma unroll
            for (int r = 0; r < 8; ++r)
                base[(size_t)(t * 16 + cl) * kL + r + hi8] = to_bf(acc[t][r]);
    } else {                             // Q (scaled) or K: [bh, l, c]
        __bf16* base = (sect == 0 ? Q : K) + (bh * kL + l0) * kHD;
        const float sc = (sect == 0) ? 0.125f : 1.0f;   // hd^-0.5
#pragma unroll
        for (int t = 0; t < 4; ++t)
#pragma unroll
            for (int r = 0; r < 8; ++r)
                base[(size_t)(r + hi8) * kHD + t * 16 + cl] = to_bf(acc[t][r] * sc);
    }
}

// ---------------------------------------------------------------------------
// Attention: one wave per (b,h,16-query tile). Scores+bias+mask -> LDS fp32,
// softmax in LDS, then P @ V via WMMA.
__global__ __launch_bounds__(32) void attn_kernel(const __bf16* __restrict__ Q,
                                                  const __bf16* __restrict__ K,
                                                  const __bf16* __restrict__ Vt,
                                                  const float* __restrict__ bias,
                                                  const unsigned char* __restrict__ mask,
                                                  float* __restrict__ Y) {
    __shared__ float s[16 * kL];                      // 64 KB score tile
    const int lane = threadIdx.x & 31;
    const int qt = blockIdx.x & 63;                   // L/16 query tiles
    const int bh = blockIdx.x >> 6;
    const int h  = bh & (kH - 1);
    const int bb = bh >> 4;
    const int q0 = qt * 16;

    const __bf16* Qp  = Q  + (size_t)bh * kL * kHD;
    const __bf16* Kp  = K  + (size_t)bh * kL * kHD;
    const __bf16* Vtp = Vt + (size_t)bh * kHD * kL;

    v16bf qa0 = load_a_frag(Qp, kHD, q0, 0, lane);
    v16bf qa1 = load_a_frag(Qp, kHD, q0, 32, lane);

    const int hi8 = (lane >> 4) << 3;
    const int cl  = lane & 15;

    // scores: S = Q·K^T + bias, masked, into LDS
    for (int kt = 0; kt < kL / 16; ++kt) {
        v8f sacc = {};
        sacc = wmma_bf16(qa0, load_b_frag(Kp, kHD, 0,  kt * 16, lane), sacc);
        sacc = wmma_bf16(qa1, load_b_frag(Kp, kHD, 32, kt * 16, lane), sacc);
        int kk = kt * 16 + cl;
        bool ok = mask[(size_t)bb * kL + kk] != 0;
        const float* bp = bias + ((size_t)(bb * kL + q0 + hi8) * kL + kk) * kH + h;
#pragma unroll
        for (int r = 0; r < 8; ++r) {
            float scv = sacc[r] + bp[(size_t)r * kL * kH];
            s[(r + hi8) * kL + kk] = ok ? scv : -3.0e38f;
        }
    }

    // fp32 softmax per row, in LDS
    for (int r = 0; r < 16; ++r) {
        float* row = s + r * kL;
        float m = -3.0e38f;
        for (int i = lane; i < kL; i += 32) m = fmaxf(m, row[i]);
#pragma unroll
        for (int off = 16; off > 0; off >>= 1) m = fmaxf(m, __shfl_xor(m, off, 32));
        float sum = 0.f;
        for (int i = lane; i < kL; i += 32) {
            float e = __expf(row[i] - m);
            row[i] = e;
            sum += e;
        }
#pragma unroll
        for (int off = 16; off > 0; off >>= 1) sum += __shfl_xor(sum, off, 32);
        float inv = 1.f / sum;
        for (int i = lane; i < kL; i += 32) row[i] *= inv;
    }

    // O = P (16x1024) @ V (1024x64); V is transposed so B-fragments are contiguous
    v8f o[4] = {};
    for (int k0 = 0; k0 < kL; k0 += 32) {
        v16bf a = load_a_frag_lds(s, kL, k0, lane);
#pragma unroll
        for (int t = 0; t < 4; ++t)
            o[t] = wmma_bf16(a, load_b_frag(Vtp, kL, k0, t * 16, lane), o[t]);
    }
#pragma unroll
    for (int t = 0; t < 4; ++t)
#pragma unroll
        for (int r = 0; r < 8; ++r)
            // 'bqhc' layout: [B, L, H, hd] == [B, L, DIM]
            Y[(size_t)(bb * kL + q0 + r + hi8) * kDIM + h * kHD + t * 16 + cl] = o[t][r];
}

// ---------------------------------------------------------------------------
// Gate: ylin = x @ W_g + b_g ; ybf = bf16( sigmoid(ylin) * y_attn )
__global__ __launch_bounds__(256) void gate_gemm(const __bf16* __restrict__ Xb,
                                                 const __bf16* __restrict__ Wgt,
                                                 const float* __restrict__ bg,
                                                 const float* __restrict__ Y,
                                                 __bf16* __restrict__ Ybf) {
    const int lane = threadIdx.x & 31;
    const int wid  = blockIdx.x * (blockDim.x >> 5) + (threadIdx.x >> 5);
    const int tilesN = kDIM / 64;                     // 16
    const int tm = wid / tilesN, tn = wid % tilesN;
    if (tm >= kROWS / 16) return;
    const int m0 = tm * 16, n0 = tn * 64;

    v8f acc[4] = {};
    for (int k0 = 0; k0 < kDIM; k0 += 32) {
        v16bf a = load_a_frag(Xb, kDIM, m0, k0, lane);
#pragma unroll
        for (int t = 0; t < 4; ++t)
            acc[t] = wmma_bf16(a, load_b_frag(Wgt, kDIM, k0, n0 + t * 16, lane), acc[t]);
    }
    const int hi8 = (lane >> 4) << 3;
    const int cl  = lane & 15;
#pragma unroll
    for (int t = 0; t < 4; ++t)
#pragma unroll
        for (int r = 0; r < 8; ++r) {
            int row = m0 + r + hi8;
            int col = n0 + t * 16 + cl;
            float lin = acc[t][r] + bg[col];
            float g = 1.f / (1.f + __expf(-lin));
            size_t idx = (size_t)row * kDIM + col;
            Ybf[idx] = to_bf(g * Y[idx]);
        }
}

// ---------------------------------------------------------------------------
// Output projection: out = ybf @ W_o + b_o  (fp32 out)
__global__ __launch_bounds__(256) void out_gemm(const __bf16* __restrict__ Ybf,
                                                const __bf16* __restrict__ Wot,
                                                const float* __restrict__ bo,
                                                float* __restrict__ Out) {
    const int lane = threadIdx.x & 31;
    const int wid  = blockIdx.x * (blockDim.x >> 5) + (threadIdx.x >> 5);
    const int tilesN = kDIM / 64;
    const int tm = wid / tilesN, tn = wid % tilesN;
    if (tm >= kROWS / 16) return;
    const int m0 = tm * 16, n0 = tn * 64;

    v8f acc[4] = {};
    for (int k0 = 0; k0 < kDIM; k0 += 32) {
        v16bf a = load_a_frag(Ybf, kDIM, m0, k0, lane);
#pragma unroll
        for (int t = 0; t < 4; ++t)
            acc[t] = wmma_bf16(a, load_b_frag(Wot, kDIM, k0, n0 + t * 16, lane), acc[t]);
    }
    const int hi8 = (lane >> 4) << 3;
    const int cl  = lane & 15;
#pragma unroll
    for (int t = 0; t < 4; ++t)
#pragma unroll
        for (int r = 0; r < 8; ++r)
            Out[(size_t)(m0 + r + hi8) * kDIM + n0 + t * 16 + cl] = acc[t][r] + bo[n0 + t * 16 + cl];
}

// ---------------------------------------------------------------------------
extern "C" void kernel_launch(void* const* d_in, const int* in_sizes, int n_in,
                              void* d_out, int out_size, void* d_ws, size_t ws_size,
                              hipStream_t stream) {
    const float* x     = (const float*)d_in[0];
    const unsigned char* mask = (const unsigned char*)d_in[1];   // jnp bool -> 1 byte
    const float* bias  = (const float*)d_in[2];
    const float* W_qkv = (const float*)d_in[3];
    const float* W_o   = (const float*)d_in[4];
    const float* b_o   = (const float*)d_in[5];
    const float* W_g   = (const float*)d_in[6];
    const float* b_g   = (const float*)d_in[7];
    float* out = (float*)d_out;

    // workspace layout (bytes)
    char* ws = (char*)d_ws;
    size_t off = 0;
    auto alloc = [&](size_t bytes) { char* p = ws + off; off += (bytes + 255) & ~(size_t)255; return p; };
    __bf16* xb     = (__bf16*)alloc((size_t)kROWS * kDIM * 2);       // 4 MB  (row-major)
    __bf16* wqkvt  = (__bf16*)alloc((size_t)kQKVN * kDIM * 2);       // 6 MB  [3072,1024]
    __bf16* wgt    = (__bf16*)alloc((size_t)kDIM * kDIM * 2);        // 2 MB  [1024,1024]
    __bf16* wot    = (__bf16*)alloc((size_t)kDIM * kDIM * 2);        // 2 MB  [1024,1024]
    __bf16* qb     = (__bf16*)alloc((size_t)kB * kH * kL * kHD * 2); // 4 MB  [B,H,L,hd]
    __bf16* kbuf   = (__bf16*)alloc((size_t)kB * kH * kL * kHD * 2); // 4 MB  [B,H,L,hd]
    __bf16* vtbuf  = (__bf16*)alloc((size_t)kB * kH * kHD * kL * 2); // 4 MB  [B,H,hd,L]
    float*  yatt   = (float*)alloc((size_t)kROWS * kDIM * 4);        // 8 MB
    __bf16* ybf    = (__bf16*)alloc((size_t)kROWS * kDIM * 2);       // 4 MB
    (void)ws_size;

    // 1. conversions: x row-major bf16; weights transposed to [N,K] bf16
    {
        int n = kROWS * kDIM;
        cvt_f32_to_bf16<<<(n + 255) / 256, 256, 0, stream>>>(x, xb, n);
        cvt_transpose_bf16<<<dim3(kQKVN / 16, kDIM / 16), 256, 0, stream>>>(W_qkv, wqkvt, kDIM, kQKVN);
        cvt_transpose_bf16<<<dim3(kDIM / 16, kDIM / 16), 256, 0, stream>>>(W_g, wgt, kDIM, kDIM);
        cvt_transpose_bf16<<<dim3(kDIM / 16, kDIM / 16), 256, 0, stream>>>(W_o, wot, kDIM, kDIM);
    }

    // 2. QKV projection: (2048/16)*(3072/64) = 6144 waves, 8 waves/block
    qkv_gemm<<<6144 / 8, 256, 0, stream>>>(xb, wqkvt, qb, kbuf, vtbuf);

    // 3. attention: B*H*(L/16) = 2048 single-wave workgroups (64 KB LDS each)
    attn_kernel<<<kB * kH * (kL / 16), 32, 0, stream>>>(qb, kbuf, vtbuf, bias, mask, yatt);

    // 4. gate: (2048/16)*(1024/64) = 2048 waves
    gate_gemm<<<2048 / 8, 256, 0, stream>>>(xb, wgt, b_g, yatt, ybf);

    // 5. output projection
    out_gemm<<<2048 / 8, 256, 0, stream>>>(ybf, wot, b_o, out);
}
